// ChannelAttention_6811818132404
// MI455X (gfx1250) — compile-verified, matrix operand order checked by reading, not run on hardware
//
#include <hip/hip_runtime.h>

#define B_ 8
#define C_ 128
#define H_ 96
#define W_ 96
#define HW_ 9216          // H*W
#define NH_ 8             // heads
#define HD_ 64            // head dim
#define NPIX_ 9216        // contraction length for gemm1
#define SPLIT_ 4          // split-K factor for gemm1
#define NCHUNK_ 36        // HW/256 pixel chunks per (b,c)

typedef __attribute__((ext_vector_type(8)))  float  v8f;
typedef __attribute__((ext_vector_type(8)))  __bf16 v8bf;
typedef __attribute__((ext_vector_type(16))) __bf16 v16bf;

// Load one wave-lane's share of a 16x32 bf16 WMMA operand tile.
// p = row base + (lane>=16 ? 8 : 0); returns elements [0..7] and [16..23],
// matching the ISA 16-bit A/B operand layout (lanes 0-15: K 0..7 & 16..23,
// lanes 16-31: K 8..15 & 24..31).
__device__ __forceinline__ v16bf load_tile16(const __bf16* p) {
    v8bf lo = *reinterpret_cast<const v8bf*>(p);
    v8bf hi = *reinterpret_cast<const v8bf*>(p + 16);
    return __builtin_shufflevector(lo, hi, 0,1,2,3,4,5,6,7,8,9,10,11,12,13,14,15);
}

// ---------------------------------------------------------------------------
// K1: depthwise qkv conv (groups=C, multiplier 12). Writes Q,K,Vrow as bf16
// in [b,head,d,n] (n contiguous -> fully coalesced stores). Also reduces the
// per-block sum-of-squares of q and k (8 row-fragments per block) so the
// L2-norm pass never has to re-read the 151MB of Q/K.
// ---------------------------------------------------------------------------
__global__ void qkv_conv_kernel(const float* __restrict__ x,
                                const float* __restrict__ w_qkv,
                                const float* __restrict__ b_qkv,
                                __bf16* __restrict__ Q,
                                __bf16* __restrict__ K,
                                __bf16* __restrict__ Vrow,
                                float* __restrict__ pp) {
    const int chunk = blockIdx.x % NCHUNK_;
    const int c     = (blockIdx.x / NCHUNK_) % C_;
    const int b     = blockIdx.x / (NCHUNK_ * C_);
    const int pix   = chunk * 256 + threadIdx.x;
    const int h = pix / W_, w = pix % W_;

    const float* xp = x + ((size_t)(b * C_ + c)) * HW_;
    float p[9];
#pragma unroll
    for (int kh = 0; kh < 3; ++kh)
#pragma unroll
        for (int kw = 0; kw < 3; ++kw) {
            int hh = h + kh - 1, ww = w + kw - 1;
            bool ok = (hh >= 0) & (hh < H_) & (ww >= 0) & (ww < W_);
            p[kh * 3 + kw] = ok ? xp[hh * W_ + ww] : 0.0f;
        }

    float sq[4] = {0.f, 0.f, 0.f, 0.f};
    float sk[4] = {0.f, 0.f, 0.f, 0.f};

#pragma unroll
    for (int j = 0; j < 12; ++j) {
        const float* wj = w_qkv + (size_t)(c * 12 + j) * 9;
        float acc = b_qkv[c * 12 + j];
#pragma unroll
        for (int i = 0; i < 9; ++i) acc += wj[i] * p[i];

        const int e = j & 3;          // expan index
        const int sel = j >> 2;       // 0=q 1=k 2=v
        const int head = 2 * e + (c >> 6);
        const int d = c & 63;
        const size_t idx = (((size_t)b * NH_ + head) * HD_ + d) * NPIX_ + pix;
        __bf16 bv = (__bf16)acc;
        if (sel == 0)      { Q[idx] = bv;   sq[e] = acc * acc; }
        else if (sel == 1) { K[idx] = bv;   sk[e] = acc * acc; }
        else               { Vrow[idx] = bv; }
    }

    // block-reduce 8 sum-of-squares fragments (q:e=0..3, k:e=0..3)
    __shared__ float lred[8][8];
    const int wave = threadIdx.x >> 5, lane = threadIdx.x & 31;
#pragma unroll
    for (int cmb = 0; cmb < 8; ++cmb) {
        float v = (cmb < 4) ? sq[cmb] : sk[cmb - 4];
#pragma unroll
        for (int s = 16; s > 0; s >>= 1) v += __shfl_xor(v, s, 32);
        if (lane == 0) lred[wave][cmb] = v;
    }
    __syncthreads();
    if (threadIdx.x < 8) {           // threadIdx.x = cmb = sel*4 + e
        float s = 0.f;
#pragma unroll
        for (int wv = 0; wv < 8; ++wv) s += lred[wv][threadIdx.x];
        pp[((size_t)(b * C_ + c) * 8 + threadIdx.x) * NCHUNK_ + chunk] = s;
    }
}

// ---------------------------------------------------------------------------
// K2: finalize inverse L2 norms from 1.1MB of per-block partials.
// One thread per row (4096 Q rows + 4096 K rows).
// ---------------------------------------------------------------------------
__global__ void normfin_kernel(const float* __restrict__ pp,
                               float* __restrict__ invq,
                               float* __restrict__ invk) {
    const int gid = blockIdx.x * 256 + threadIdx.x;     // 0..8191
    const bool isQ = gid < (B_ * NH_ * HD_);
    const int r = isQ ? gid : gid - B_ * NH_ * HD_;     // bh*64+d
    const int bh = r >> 6, d = r & 63;
    const int b = bh >> 3, head = bh & 7;
    const int e = head >> 1;
    const int c = (head & 1) * 64 + d;
    const float* base = pp +
        ((size_t)(b * C_ + c) * 8 + (isQ ? 0 : 4) + e) * NCHUNK_;
    float s = 0.f;
#pragma unroll
    for (int i = 0; i < NCHUNK_; ++i) s += base[i];
    float inv = 1.0f / fmaxf(sqrtf(s), 1e-12f);
    (isQ ? invq : invk)[r] = inv;
}

// ---------------------------------------------------------------------------
// K1b: V transpose Vrow [bh][d][n] -> Vt [bh][n][d] via 64x64 LDS tiles.
// Row stride 72 bf16 (144B, 16B-aligned) keeps b128 LDS accesses legal.
// Coalesced global reads AND writes (vs 16x sector-amplified direct scatter).
// ---------------------------------------------------------------------------
__global__ void vtrans_kernel(const __bf16* __restrict__ Vrow,
                              __bf16* __restrict__ Vt) {
    const int ntile = NPIX_ / 64;                 // 144
    const int bh = blockIdx.x / ntile;
    const int n0 = (blockIdx.x % ntile) * 64;
    __shared__ __bf16 lds[64][72];

    const __bf16* src = Vrow + (size_t)bh * HD_ * NPIX_ + n0;
#pragma unroll
    for (int i = 0; i < 2; ++i) {
        int g = threadIdx.x + i * 256;
        int d = g >> 3, n8 = (g & 7) * 8;
        v8bf v = *reinterpret_cast<const v8bf*>(src + (size_t)d * NPIX_ + n8);
        *reinterpret_cast<v8bf*>(&lds[d][n8]) = v;
    }
    __syncthreads();
#pragma unroll
    for (int i = 0; i < 2; ++i) {
        int g = threadIdx.x + i * 256;
        int n = g >> 3, d8 = (g & 7) * 8;
        v8bf o;
#pragma unroll
        for (int j = 0; j < 8; ++j) o[j] = lds[d8 + j][n];
        *reinterpret_cast<v8bf*>(Vt + ((size_t)bh * NPIX_ + n0 + n) * HD_ + d8) = o;
    }
}

// ---------------------------------------------------------------------------
// K3: split-K partials of attn_raw = Q @ K^T. (B*NH)*SPLIT blocks; raw f32
// partials into an L2-resident 4MB buffer. Wave = one d-tile x two e-tiles.
// ---------------------------------------------------------------------------
__global__ void gemm_qkT_partial_kernel(const __bf16* __restrict__ Q,
                                        const __bf16* __restrict__ K,
                                        float* __restrict__ part) {
    const int bh = blockIdx.x / SPLIT_;          // 0..63
    const int sp = blockIdx.x % SPLIT_;          // 0..3
    const int wave = threadIdx.x >> 5;
    const int lane = threadIdx.x & 31;
    const int m = lane & 15, hi = lane >> 4;
    const int dt  = wave >> 1;                   // 0..3
    const int et0 = (wave & 1) * 2;              // 0 or 2

    const __bf16* pa  = Q + ((size_t)bh * HD_ + dt * 16 + m) * NPIX_ + hi * 8;
    const __bf16* pb0 = K + ((size_t)bh * HD_ + et0 * 16 + m) * NPIX_ + hi * 8;
    const __bf16* pb1 = pb0 + (size_t)16 * NPIX_;

    const int nbeg = sp * (NPIX_ / SPLIT_);
    const int nend = nbeg + (NPIX_ / SPLIT_);

    v8f c0 = {}, c1 = {};
    for (int n0 = nbeg; n0 < nend; n0 += 32) {
        v16bf A  = load_tile16(pa + n0);
        v16bf B0 = load_tile16(pb0 + n0);
        v16bf B1 = load_tile16(pb1 + n0);
        c0 = __builtin_amdgcn_wmma_f32_16x16x32_bf16(false, A, false, B0,
                                                     (short)0, c0, false, false);
        c1 = __builtin_amdgcn_wmma_f32_16x16x32_bf16(false, A, false, B1,
                                                     (short)0, c1, false, false);
    }

    float* base = part + (((size_t)sp * (B_ * NH_) + bh) * HD_) * HD_;
#pragma unroll
    for (int r = 0; r < 8; ++r) {
        const int d  = dt * 16 + r + 8 * hi;
        const int e0 = et0 * 16 + m;
        base[(size_t)d * HD_ + e0]      = c0[r];
        base[(size_t)d * HD_ + e0 + 16] = c1[r];
    }
}

// ---------------------------------------------------------------------------
// K4: combine split-K partials + fold normalization scales + temperature,
// then softmax over rows of 64; one wave (32 lanes x 2 elems) per row.
// ---------------------------------------------------------------------------
__global__ void softmax_kernel(const float* __restrict__ part,
                               const float* __restrict__ invq,
                               const float* __restrict__ invk,
                               const float* __restrict__ t,
                               __bf16* __restrict__ attnb) {
    const int row  = blockIdx.x * 8 + (threadIdx.x >> 5);  // bh*64+d
    const int lane = threadIdx.x & 31;
    const int bh = row >> 6;
    const int d  = row & 63;

    float s0 = 0.0f, s1 = 0.0f;
#pragma unroll
    for (int sp = 0; sp < SPLIT_; ++sp) {
        const float* p = part +
            (((size_t)sp * (B_ * NH_) + bh) * HD_ + d) * HD_;
        s0 += p[lane];
        s1 += p[lane + 32];
    }
    const float sc = invq[row] * t[bh & (NH_ - 1)];
    float a0 = s0 * sc * invk[bh * HD_ + lane];
    float a1 = s1 * sc * invk[bh * HD_ + lane + 32];

    float mx = fmaxf(a0, a1);
#pragma unroll
    for (int s = 16; s > 0; s >>= 1) mx = fmaxf(mx, __shfl_xor(mx, s, 32));
    float e0 = __expf(a0 - mx), e1 = __expf(a1 - mx);
    float sum = e0 + e1;
#pragma unroll
    for (int s = 16; s > 0; s >>= 1) sum += __shfl_xor(sum, s, 32);
    float rs = 1.0f / sum;
    attnb[(size_t)row * HD_ + lane]      = (__bf16)(e0 * rs);
    attnb[(size_t)row * HD_ + lane + 32] = (__bf16)(e1 * rs);
}

// ---------------------------------------------------------------------------
// K5: O = attn @ V per (b,head): 64 x 9216, K=64 (two k-steps of 32).
// Block = 8 waves covers a 64(d) x 32(n) output slab; V read from the
// n-major/e-contiguous Vt layout so B tiles are contiguous 16-B loads.
// ---------------------------------------------------------------------------
__global__ void gemm_av_kernel(const __bf16* __restrict__ attnb,
                               const __bf16* __restrict__ Vt,
                               float* __restrict__ O) {
    const int nblk = NPIX_ / 32;                 // 288
    const int bh = blockIdx.x / nblk;
    const int nc = blockIdx.x % nblk;
    const int wave = threadIdx.x >> 5;
    const int lane = threadIdx.x & 31;
    const int m = lane & 15, hi = lane >> 4;
    const int dt = wave >> 1, nt = wave & 1;

    const __bf16* pa = attnb + ((size_t)bh * HD_ + dt * 16 + m) * HD_ + hi * 8;
    const int col = nc * 32 + nt * 16 + m;
    const __bf16* pb = Vt + ((size_t)bh * NPIX_ + col) * HD_ + hi * 8;

    v8f c = {};
#pragma unroll
    for (int kk = 0; kk < HD_; kk += 32) {
        v16bf A  = load_tile16(pa + kk);
        v16bf Bt = load_tile16(pb + kk);
        c = __builtin_amdgcn_wmma_f32_16x16x32_bf16(false, A, false, Bt,
                                                    (short)0, c, false, false);
    }

    float* po = O + ((size_t)bh * HD_ + dt * 16) * NPIX_;
#pragma unroll
    for (int r = 0; r < 8; ++r) {
        const int d = r + 8 * hi;
        po[(size_t)d * NPIX_ + col] = c[r];
    }
}

// ---------------------------------------------------------------------------
// K6: fusion conv: groups=C, 4 in-channels per group, 3x3. Input channel
// c*4+e maps to O plane [b, head=2e+(c>>6), d=c&63].
// ---------------------------------------------------------------------------
__global__ void fus_conv_kernel(const float* __restrict__ O,
                                const float* __restrict__ w_fus,
                                const float* __restrict__ b_fus,
                                float* __restrict__ out) {
    const int chunk = blockIdx.x % NCHUNK_;
    const int c     = (blockIdx.x / NCHUNK_) % C_;
    const int b     = blockIdx.x / (NCHUNK_ * C_);
    const int pix   = chunk * 256 + threadIdx.x;
    const int h = pix / W_, w = pix % W_;

    float acc = b_fus[c];
#pragma unroll
    for (int e = 0; e < 4; ++e) {
        const int head = 2 * e + (c >> 6);
        const int d = c & 63;
        const float* plane = O + ((size_t)(b * NH_ + head) * HD_ + d) * NPIX_;
        const float* wf = w_fus + (size_t)(c * 4 + e) * 9;
#pragma unroll
        for (int kh = 0; kh < 3; ++kh)
#pragma unroll
            for (int kw = 0; kw < 3; ++kw) {
                int hh = h + kh - 1, ww = w + kw - 1;
                bool ok = (hh >= 0) & (hh < H_) & (ww >= 0) & (ww < W_);
                if (ok) acc += wf[kh * 3 + kw] * plane[hh * W_ + ww];
            }
    }
    out[((size_t)(b * C_ + c)) * HW_ + pix] = acc;
}

// ---------------------------------------------------------------------------
extern "C" void kernel_launch(void* const* d_in, const int* in_sizes, int n_in,
                              void* d_out, int out_size, void* d_ws, size_t ws_size,
                              hipStream_t stream) {
    (void)in_sizes; (void)n_in; (void)out_size; (void)ws_size;

    const float* x     = (const float*)d_in[0];
    const float* w_qkv = (const float*)d_in[1];
    const float* b_qkv = (const float*)d_in[2];
    const float* t     = (const float*)d_in[3];
    const float* w_fus = (const float*)d_in[4];
    const float* b_fus = (const float*)d_in[5];
    float* out = (float*)d_out;

    const size_t QE = (size_t)B_ * NH_ * HD_ * NPIX_;   // 37,748,736 elems
    char* ws = (char*)d_ws;
    size_t off = 0;
    __bf16* Q    = (__bf16*)(ws + off); off += QE * 2;
    __bf16* K    = (__bf16*)(ws + off); off += QE * 2;
    __bf16* Vrow = (__bf16*)(ws + off); off += QE * 2;
    __bf16* Vt   = (__bf16*)(ws + off); off += QE * 2;
    float* invq  = (float*)(ws + off); off += (size_t)B_ * NH_ * HD_ * 4;
    float* invk  = (float*)(ws + off); off += (size_t)B_ * NH_ * HD_ * 4;
    float* pp    = (float*)(ws + off); off += (size_t)B_ * C_ * 8 * NCHUNK_ * 4;
    float* part  = (float*)(ws + off); off += (size_t)SPLIT_ * B_ * NH_ * HD_ * HD_ * 4;
    __bf16* attnb = (__bf16*)(ws + off); off += (size_t)B_ * NH_ * HD_ * HD_ * 2;
    float* O = (float*)(ws + off); off += QE * 4;

    dim3 blk(256);
    qkv_conv_kernel<<<dim3(B_ * C_ * NCHUNK_), blk, 0, stream>>>(
        x, w_qkv, b_qkv, Q, K, Vrow, pp);
    normfin_kernel<<<dim3(2 * B_ * NH_ * HD_ / 256), blk, 0, stream>>>(
        pp, invq, invk);
    vtrans_kernel<<<dim3(B_ * NH_ * (NPIX_ / 64)), blk, 0, stream>>>(
        Vrow, Vt);
    gemm_qkT_partial_kernel<<<dim3(B_ * NH_ * SPLIT_), blk, 0, stream>>>(
        Q, K, part);
    softmax_kernel<<<dim3(B_ * NH_ * HD_ / 8), blk, 0, stream>>>(
        part, invq, invk, t, attnb);
    gemm_av_kernel<<<dim3(B_ * NH_ * (NPIX_ / 32)), blk, 0, stream>>>(
        attnb, Vt, O);
    fus_conv_kernel<<<dim3(B_ * C_ * NCHUNK_), blk, 0, stream>>>(
        O, w_fus, b_fus, out);
}
